// PanODE_80848464380373
// MI455X (gfx1250) — matrix operand order repaired and checked by reading, not compiled, not run on hardware
//
#include <hip/hip_runtime.h>
#include <hip/hip_bf16.h>

// ---------------------------------------------------------------------------
// PAN (Chebyshev-spectral) ODE solver for MI455X (gfx1250).
// Dominant cost: 60x [12288x512]x[512x512] GEMMs -> v_wmma_f32_16x16x32_f16,
// tiles staged via Tensor Data Mover (tensor_load_to_lds + s_wait_tensorcnt),
// double-buffered LDS with TDM-generated bank-conflict padding.
// ---------------------------------------------------------------------------

typedef __attribute__((ext_vector_type(16))) _Float16 v16h;
typedef __attribute__((ext_vector_type(8)))  _Float16 v8h;
typedef __attribute__((ext_vector_type(8)))  float    v8f;
typedef __attribute__((ext_vector_type(4)))  unsigned u32x4;
typedef __attribute__((ext_vector_type(8)))  int      i32x8;
typedef __attribute__((ext_vector_type(4)))  int      i32x4;

#define PAN_C   24      // Chebyshev coefficients
#define PAN_P   48      // collocation nodes
#define PAN_B   256     // batch
#define PAN_D   512     // dim
#define PAN_BD  (PAN_B * PAN_D)          // 131072
#define PAN_M   (PAN_P * PAN_B)          // 12288 GEMM rows
#define PAN_PI  3.14159265358979323846f

// LDS tile row stride: 32 halves (16 dw) + 4 dw TDM pad = 20 dw = 40 halves.
// 16-byte aligned rows; lane->row stride 20 dwords hits 16 distinct banks.
#define PAN_LDS_STRIDE 40

#if defined(__has_builtin)
#if __has_builtin(__builtin_amdgcn_tensor_load_to_lds)
#define PAN_TDM 1
#endif
#endif

// ---------------------------------------------------------------------------
// One-time prep kernels
// ---------------------------------------------------------------------------
__global__ __launch_bounds__(256) void pan_prep_weights(
    const float* __restrict__ W1, const float* __restrict__ W2,
    _Float16* __restrict__ W1T, _Float16* __restrict__ W2T) {
  int i = blockIdx.x * 256 + threadIdx.x;          // over 512*512
  int k = i >> 9, n = i & 511;
  W1T[(size_t)n * 512 + k] = (_Float16)W1[i];
  W2T[(size_t)n * 512 + k] = (_Float16)W2[i];
}

__global__ __launch_bounds__(256) void pan_prep_cheb(
    float* __restrict__ T, float* __restrict__ V) {
  int i = blockIdx.x * 256 + threadIdx.x;          // over 24*48
  if (i < PAN_C * PAN_P) {
    int c = i / PAN_P, p = i % PAN_P;
    float ang = PAN_PI * ((float)p + 0.5f) / (float)PAN_P;
    float t = __cosf((float)c * ang);
    T[i] = t;
    float v = (2.0f / (float)PAN_P) * t;
    if (c == 0) v *= 0.5f;
    V[i] = v;
  }
}

__global__ __launch_bounds__(256) void pan_head(
    const float* __restrict__ t_span, const float* __restrict__ y_init,
    float* __restrict__ out, float* __restrict__ y0) {
  int i = blockIdx.x * 256 + threadIdx.x;
  if (i < 7) out[i] = t_span[i];
  if (i < PAN_BD) {
    float v = y_init[i];
    out[7 + i] = v;
    y0[i] = v;
  }
}

__global__ __launch_bounds__(256) void pan_initB(
    const float* __restrict__ y0, float* __restrict__ Bc) {
  int tid = blockIdx.x * 256 + threadIdx.x;        // (b,d)
  size_t base = (size_t)tid * PAN_C;
  Bc[base] = y0[tid];
#pragma unroll
  for (int n = 1; n < PAN_C; n++) Bc[base + n] = 0.0f;
}

// Chebyshev synthesis: y_nodes[p*B + b][d] (f16) = sum_c B[b,d,c] * T[c,p]
__global__ __launch_bounds__(256) void pan_nodes(
    const float* __restrict__ Bc, const float* __restrict__ T,
    _Float16* __restrict__ ynodes) {
  __shared__ float Ts[PAN_C * PAN_P];
  for (int i = threadIdx.x; i < PAN_C * PAN_P; i += 256) Ts[i] = T[i];
  __syncthreads();
  int tid = blockIdx.x * 256 + threadIdx.x;        // tid = b*512 + d
  float c[PAN_C];
  size_t base = (size_t)tid * PAN_C;
#pragma unroll
  for (int j = 0; j < PAN_C; j++) c[j] = Bc[base + j];
#pragma unroll 4
  for (int p = 0; p < PAN_P; p++) {
    float s = 0.0f;
#pragma unroll
    for (int j = 0; j < PAN_C; j++) s = fmaf(c[j], Ts[j * PAN_P + p], s);
    ynodes[(size_t)p * PAN_BD + tid] = (_Float16)s;  // row p*256+b, col d
  }
}

// ---------------------------------------------------------------------------
// TDM helpers: 2D tile DMA (32 elems x 128 rows, 2B elems, row stride 512)
// into padded LDS (pad_interval=3: every 16 dwords, pad_amount=3: +4 dwords).
// D# packing per CDNA5 ISA.
// ---------------------------------------------------------------------------
__device__ inline unsigned pan_lds_off(const void* p) {
  return (unsigned)(unsigned long long)(uintptr_t)p;
}

#ifdef PAN_TDM
__device__ inline void pan_tdm_load_tile(const _Float16* gtile, unsigned lds_off,
                                         int rows_total) {
  unsigned long long ga = (unsigned long long)(uintptr_t)gtile;
  u32x4 g0 = {
      1u,                                            // count=1 (valid user D#)
      lds_off,                                       // lds_addr (bytes)
      (unsigned)ga,                                  // global_addr[31:0]
      ((unsigned)(ga >> 32) & 0x1FFFFFFu) | (2u << 30)  // addr[56:32] | type=2
  };
  i32x8 g1 = {
      (int)((1u << 16) | (1u << 20) | (3u << 22) | (3u << 25)),
      //    data_size=2B | pad_en    | pad_int=16dw | pad_amt=4dw
      (int)(512u << 16),                             // tensor_dim0[15:0]=512
      (int)(((unsigned)rows_total & 0xFFFFu) << 16), // dim0 hi=0 | tensor_dim1 lo
      (int)((((unsigned)rows_total >> 16) & 0xFFFFu) | (32u << 16)), // tile_dim0=32
      (int)128,                                      // tile_dim1=128, tile_dim2=0
      (int)512,                                      // tensor_dim0_stride=512
      0, 0
  };
  i32x4 z4 = {0, 0, 0, 0};
  i32x8 z8 = {0, 0, 0, 0, 0, 0, 0, 0};
  __builtin_amdgcn_tensor_load_to_lds(g0, g1, z4, z4, z8, 0);
}
#endif

// ---------------------------------------------------------------------------
// WMMA GEMM: C[M x 512] = A[M x 512](f16) * Wt(f16, [n][k]) + bias
// 128x128 tile/WG; 8 waves in 4x2 grid, each wave: 32 rows x 64 cols
// (2 M-tiles x 4 N-tiles -> 12 ds_load_b128 per 8 WMMAs). BK=32, 16 K-steps,
// double-buffered LDS fed by TDM, overlapped via TENSORcnt.
// ---------------------------------------------------------------------------
__device__ inline v16h pan_join(v8h lo, v8h hi) {
  return __builtin_shufflevector(lo, hi, 0, 1, 2, 3, 4, 5, 6, 7,
                                 8, 9, 10, 11, 12, 13, 14, 15);
}

template <bool TANH, bool OUT_F16>
__global__ __launch_bounds__(256) void pan_gemm(
    const _Float16* __restrict__ A,    // [M x 512]
    const _Float16* __restrict__ Bt,   // [512 x 512] transposed weights
    const float* __restrict__ bias,    // [512]
    _Float16* __restrict__ outH,       // if OUT_F16
    float* __restrict__ outF,          // else
    int M) {
  __shared__ __align__(16) _Float16 As[2][128 * PAN_LDS_STRIDE];
  __shared__ __align__(16) _Float16 Bs[2][128 * PAN_LDS_STRIDE];

  const int tid  = threadIdx.x;
  const int lane = tid & 31;
  const int wave = tid >> 5;           // 0..7
  const int hi   = lane >> 4;          // 0/1
  const int l16  = lane & 15;
  const int wr   = wave & 3;           // wave row group: 32 rows each
  const int wc   = wave >> 2;          // wave col group: 64 cols each
  const int rowBase = blockIdx.x * 128;
  const int colBase = blockIdx.y * 128;

  v8f acc[8];
#pragma unroll
  for (int i = 0; i < 8; i++) acc[i] = (v8f){0, 0, 0, 0, 0, 0, 0, 0};

#ifdef PAN_TDM
  const bool issuer = (wave == 0);
  if (issuer) {
    pan_tdm_load_tile(A + (size_t)rowBase * 512, pan_lds_off(&As[0][0]), M);
    pan_tdm_load_tile(Bt + (size_t)colBase * 512, pan_lds_off(&Bs[0][0]), 512);
    __builtin_amdgcn_s_wait_tensorcnt(0);
  }
  __syncthreads();
#endif

  for (int step = 0; step < 16; ++step) {
    const int cur = step & 1;
#ifdef PAN_TDM
    // Kick DMA for the next tile into the alternate buffer; it runs while
    // this buffer feeds the WMMAs below.
    if (issuer && step + 1 < 16) {
      const int nxt = cur ^ 1;
      pan_tdm_load_tile(A + (size_t)rowBase * 512 + (step + 1) * 32,
                        pan_lds_off(&As[nxt][0]), M);
      pan_tdm_load_tile(Bt + (size_t)colBase * 512 + (step + 1) * 32,
                        pan_lds_off(&Bs[nxt][0]), 512);
    }
#else
    // Fallback: cooperative vector loads into padded LDS.
    {
      int k0 = step * 32;
#pragma unroll
      for (int c = tid; c < 512; c += 256) {
        int r = c >> 2, ch = c & 3;
        *(uint4*)(&As[cur][r * PAN_LDS_STRIDE + ch * 8]) =
            *(const uint4*)(A + (size_t)(rowBase + r) * 512 + k0 + ch * 8);
        *(uint4*)(&Bs[cur][r * PAN_LDS_STRIDE + ch * 8]) =
            *(const uint4*)(Bt + (size_t)(colBase + r) * 512 + k0 + ch * 8);
      }
      __syncthreads();
    }
#endif

    // A fragments for this wave's two 16-row strips
    v16h af[2];
#pragma unroll
    for (int mt = 0; mt < 2; mt++) {
      const _Float16* ap = &As[cur][(wr * 32 + mt * 16 + l16) * PAN_LDS_STRIDE];
      af[mt] = pan_join(*(const v8h*)(ap + hi * 8), *(const v8h*)(ap + 16 + hi * 8));
    }
#pragma unroll
    for (int nt = 0; nt < 4; nt++) {
      const _Float16* bp =
          &Bs[cur][(wc * 64 + nt * 16 + l16) * PAN_LDS_STRIDE];
      v16h bf = pan_join(*(const v8h*)(bp + hi * 8), *(const v8h*)(bp + 16 + hi * 8));
#pragma unroll
      for (int mt = 0; mt < 2; mt++) {
        acc[mt * 4 + nt] = __builtin_amdgcn_wmma_f32_16x16x32_f16(
            false, af[mt], false, bf, (short)0, acc[mt * 4 + nt], false, false);
      }
    }

#ifdef PAN_TDM
    if (issuer) __builtin_amdgcn_s_wait_tensorcnt(0);
#endif
    __syncthreads();
  }

  // Epilogue: D vgpr i -> row = hi*8 + i, col = lane%16 within each subtile
#pragma unroll
  for (int nt = 0; nt < 4; nt++) {
    int col = colBase + wc * 64 + nt * 16 + l16;
    float bv = bias[col];
#pragma unroll
    for (int mt = 0; mt < 2; mt++) {
#pragma unroll
      for (int i = 0; i < 8; i++) {
        int row = rowBase + wr * 32 + mt * 16 + hi * 8 + i;
        float v = acc[mt * 4 + nt][i] + bv;
        if (TANH) v = tanhf(v);
        if (OUT_F16) outH[(size_t)row * 512 + col] = (_Float16)v;
        else         outF[(size_t)row * 512 + col] = v;
      }
    }
  }
}

// ---------------------------------------------------------------------------
// Projection + antiderivative + Picard update
// ---------------------------------------------------------------------------
__global__ __launch_bounds__(256) void pan_project(
    const float* __restrict__ F,       // [48 * 131072], row p*131072 + tid
    const float* __restrict__ V,       // [24 * 48]
    const float* __restrict__ t_span, int interval,
    const float* __restrict__ y0, float* __restrict__ Bc) {
  __shared__ float Vs[PAN_C * PAN_P];
  for (int i = threadIdx.x; i < PAN_C * PAN_P; i += 256) Vs[i] = V[i];
  __syncthreads();
  int tid = blockIdx.x * 256 + threadIdx.x;        // (b,d)

  float fv[PAN_P];
#pragma unroll 8
  for (int p = 0; p < PAN_P; p++) fv[p] = F[(size_t)p * PAN_BD + tid];

  float a[PAN_C + 2];
#pragma unroll 4
  for (int c = 0; c < PAN_C; c++) {
    float s = 0.0f;
#pragma unroll
    for (int p = 0; p < PAN_P; p++) s = fmaf(fv[p], Vs[c * PAN_P + p], s);
    a[c] = s;
  }
  a[PAN_C] = 0.0f;
  a[PAN_C + 1] = 0.0f;

  float scale = 0.5f * (t_span[interval + 1] - t_span[interval]);
  size_t base = (size_t)tid * PAN_C;
  float b0 = 0.0f;
#pragma unroll
  for (int n = 1; n < PAN_C; n++) {
    float r = (a[n - 1] - a[n + 1]) / (2.0f * (float)n);
    b0 += (n & 1) ? r : -r;                        // -sum r*(-1)^n
    Bc[base + n] = scale * r;
  }
  Bc[base] = scale * b0 + y0[tid];
}

__global__ __launch_bounds__(256) void pan_final(
    const float* __restrict__ Bc, float* __restrict__ y0,
    float* __restrict__ traj_out) {
  int tid = blockIdx.x * 256 + threadIdx.x;
  size_t base = (size_t)tid * PAN_C;
  float s = 0.0f;
#pragma unroll
  for (int c = 0; c < PAN_C; c++) s += Bc[base + c];
  y0[tid] = s;
  traj_out[tid] = s;
}

// ---------------------------------------------------------------------------
// Host orchestration (graph-capture safe: only kernel launches on stream)
// ---------------------------------------------------------------------------
extern "C" void kernel_launch(void* const* d_in, const int* in_sizes, int n_in,
                              void* d_out, int out_size, void* d_ws, size_t ws_size,
                              hipStream_t stream) {
  const float* y_init = (const float*)d_in[0];     // [256*512]
  const float* t_span = (const float*)d_in[1];     // [7]
  const float* W1     = (const float*)d_in[2];     // [512*512]
  const float* b1     = (const float*)d_in[3];     // [512]
  const float* W2     = (const float*)d_in[4];     // [512*512]
  const float* b2     = (const float*)d_in[5];     // [512]
  float* out          = (float*)d_out;             // [7 + 7*131072]

  char* base = (char*)d_ws;
  size_t off = 0;
  auto carve = [&](size_t bytes) -> char* {
    off = (off + 255) & ~(size_t)255;
    char* p = base + off;
    off += bytes;
    return p;
  };
  _Float16* W1T  = (_Float16*)carve((size_t)512 * 512 * 2);
  _Float16* W2T  = (_Float16*)carve((size_t)512 * 512 * 2);
  float*    Tm   = (float*)   carve((size_t)PAN_C * PAN_P * 4);
  float*    Vm   = (float*)   carve((size_t)PAN_C * PAN_P * 4);
  float*    Bc   = (float*)   carve((size_t)PAN_BD * PAN_C * 4);
  _Float16* ynd  = (_Float16*)carve((size_t)PAN_M * 512 * 2);
  _Float16* hbuf = (_Float16*)carve((size_t)PAN_M * 512 * 2);
  float*    fbuf = (float*)   carve((size_t)PAN_M * 512 * 4);
  float*    y0   = (float*)   carve((size_t)PAN_BD * 4);
  (void)ws_size; (void)in_sizes; (void)n_in; (void)out_size;

  const int nBD  = PAN_BD / 256;                   // 512 blocks
  const dim3 gemmGrid(PAN_M / 128, 512 / 128, 1);  // 96 x 4

  pan_prep_weights<<<(512 * 512) / 256, 256, 0, stream>>>(W1, W2, W1T, W2T);
  pan_prep_cheb<<<(PAN_C * PAN_P + 255) / 256, 256, 0, stream>>>(Tm, Vm);
  pan_head<<<nBD, 256, 0, stream>>>(t_span, y_init, out, y0);

  for (int iv = 0; iv < 6; iv++) {
    pan_initB<<<nBD, 256, 0, stream>>>(y0, Bc);
    for (int it = 0; it < 5; it++) {
      pan_nodes<<<nBD, 256, 0, stream>>>(Bc, Tm, ynd);
      pan_gemm<true, true><<<gemmGrid, 256, 0, stream>>>(
          ynd, W1T, b1, hbuf, (float*)nullptr, PAN_M);
      pan_gemm<false, false><<<gemmGrid, 256, 0, stream>>>(
          hbuf, W2T, b2, (_Float16*)nullptr, fbuf, PAN_M);
      pan_project<<<nBD, 256, 0, stream>>>(fbuf, Vm, t_span, iv, y0, Bc);
    }
    pan_final<<<nBD, 256, 0, stream>>>(Bc, y0, out + 7 + (size_t)(iv + 1) * PAN_BD);
  }
}
